// ViT_27350351740982
// MI455X (gfx1250) — matrix operand order repaired and checked by reading, not compiled, not run on hardware
//
#include <hip/hip_runtime.h>
#include <cstdint>
#include <cstddef>

// ---------------------------------------------------------------- types
typedef __bf16 bf16_t;
typedef __attribute__((ext_vector_type(16))) __bf16 v16bf;
typedef __attribute__((ext_vector_type(8)))  __bf16 v8bf;
typedef __attribute__((ext_vector_type(8)))  float   v8f;

// ---------------------------------------------------------------- sizes
constexpr int kE = 256, kN = 257, kB = 64, kH = 8, kHD = 32, kDepth = 12;
constexpr int kM = kB * kN;            // 16448 token rows
constexpr int kFF = 1024;
constexpr int kQRows = 272;            // 17*16 padded query rows
constexpr int kKRows = 288;            // 9*32 padded key rows
constexpr int kU = 1000, kI = 1000, kF = 128;

// ---------------------------------------------------------------- helpers
__device__ inline v8f wmma_bf16(v16bf a, v16bf b, v8f c) {
  return __builtin_amdgcn_wmma_f32_16x16x32_bf16(false, a, false, b, (short)0, c,
                                                 false, false);
}
__device__ inline v16bf cat8(v8bf lo, v8bf hi) {
  v16bf r;
#pragma unroll
  for (int i = 0; i < 8; ++i) { r[i] = lo[i]; r[i + 8] = hi[i]; }
  return r;
}
// CDNA5 async copy global->LDS (GV mode), tracked by ASYNCcnt.
__device__ inline void async_ld_b128(unsigned lds_off, const void* gaddr) {
  unsigned long long ga = (unsigned long long)gaddr;
  asm volatile("global_load_async_to_lds_b128 %0, %1, off"
               :: "v"(lds_off), "v"(ga) : "memory");
}
__device__ inline void wait_async0() {
  asm volatile("s_wait_asynccnt 0" ::: "memory");
}
// low 32 bits of a generic shared pointer == LDS byte offset (flat aperture rule)
__device__ inline unsigned lds_off_of(const void* p) {
  return (unsigned)(uintptr_t)p;
}

__device__ inline float block_sum256(float v, float* sm) {
  const int lane = threadIdx.x & 31, w = threadIdx.x >> 5;
#pragma unroll
  for (int o = 16; o; o >>= 1) v += __shfl_xor(v, o, 32);
  __syncthreads();
  if (lane == 0) sm[w] = v;
  __syncthreads();
  if (w == 0) {
    float r = (lane < 8) ? sm[lane] : 0.f;
#pragma unroll
    for (int o = 4; o; o >>= 1) r += __shfl_xor(r, o, 32);
    if (lane == 0) sm[0] = r;
  }
  __syncthreads();
  float res = sm[0];
  __syncthreads();
  return res;
}

// ---------------------------------------------------------------- kernels
// fp32 [d][k][n] -> bf16 [d][n][k]  (transposed weights for B-operand loads)
__global__ __launch_bounds__(256) void k_transp(const float* __restrict__ src,
                                                bf16_t* __restrict__ dst,
                                                int K, int Nc, long sStride, long dStride) {
  long d = blockIdx.y;
  long i = (long)blockIdx.x * 256 + threadIdx.x;
  if (i >= (long)K * Nc) return;
  int n = (int)(i / K);
  int k = (int)(i % K);
  dst[d * dStride + i] = (bf16_t)src[d * sStride + (long)k * Nc + n];
}

// build x[b][n][e] = concat(cls, user_emb, item_emb) + positions
__global__ __launch_bounds__(256) void k_embed(const int* __restrict__ user,
                                               const int* __restrict__ item,
                                               const float* __restrict__ utab,
                                               const float* __restrict__ itab,
                                               const float* __restrict__ cls,
                                               const float* __restrict__ pos,
                                               float* __restrict__ x) {
  long i = (long)blockIdx.x * 256 + threadIdx.x;   // kM*kE threads exactly
  int e = (int)(i & 255);
  long rn = i >> 8;
  int n = (int)(rn % kN);
  int b = (int)(rn / kN);
  float v;
  if (n == 0)       { v = cls[e]; }
  else if (n <= kF) { int f = n - 1;      int u  = user[b]; v = utab[((long)e * kU + u ) * kF + f]; }
  else              { int f = n - 1 - kF; int it = item[b]; v = itab[((long)e * kI + it) * kF + f]; }
  x[i] = v + pos[(long)n * kE + e];
}

// LayerNorm over E=256, one row per 256-thread block, bf16 out
__global__ __launch_bounds__(256) void k_ln(const float* __restrict__ x,
                                            const float* __restrict__ g,
                                            const float* __restrict__ be,
                                            bf16_t* __restrict__ out, int M) {
  __shared__ float sm[8];
  long row = blockIdx.x;
  int e = threadIdx.x;
  float v = x[row * kE + e];
  float s1 = block_sum256(v, sm);
  float s2 = block_sum256(v * v, sm);
  float mean = s1 * (1.f / 256.f);
  float var  = s2 * (1.f / 256.f) - mean * mean;
  float y = (v - mean) * rsqrtf(var + 1e-5f) * g[e] + be[e];
  out[row * kE + e] = (bf16_t)y;
}

// epilogue modes
#define EPI_QK    0
#define EPI_VT    1
#define EPI_RESID 2
#define EPI_GELU  3

// C[M,Nc] = A_bf16[M,K] * Wt_bf16[Nc,K]^T + bias
// block tile 128x64, K step 32, double-buffered async global->LDS weight staging
template <int EPI>
__global__ __launch_bounds__(256) void k_gemm(const bf16_t* __restrict__ A,
                                              const bf16_t* __restrict__ Wt,
                                              const float* __restrict__ bias,
                                              int M, int K, int Nc,
                                              float* __restrict__ fout,      // RESID (x, in/out)
                                              bf16_t* __restrict__ bout,     // GELU out
                                              bf16_t* __restrict__ qkout, int qkRows,
                                              bf16_t* __restrict__ vtout) {
  __shared__ __align__(32) bf16_t tileB[2][64 * 32];   // [buf][col][k] bf16, 2x4KB
  const int wave = threadIdx.x >> 5, lane = threadIdx.x & 31;
  const int lq = lane & 15, hf = lane >> 4;
  const int m0 = blockIdx.x * 128 + wave * 16;
  const int n0 = blockIdx.y * 64;
  const int colA = threadIdx.x >> 2;                // 0..63
  const int chk  = threadIdx.x & 3;                 // 0..3
  v8f acc[4] = {};

  // per-thread staging addresses (one b128 per lane per K-step)
  const bf16_t* gstage = Wt + (size_t)(n0 + colA) * K + chk * 8;
  const unsigned loff0 = lds_off_of(&tileB[0][colA * 32 + chk * 8]);
  const unsigned loff1 = lds_off_of(&tileB[1][colA * 32 + chk * 8]);

  const int nsteps = K >> 5;
  // prologue: stage tile 0
  async_ld_b128(loff0, gstage);
  wait_async0();
  __syncthreads();

  int mrow = m0 + lq; if (mrow >= M) mrow = M - 1;   // clamp, no EXEC change
  const bf16_t* arow = A + (size_t)mrow * K;

  for (int s = 0; s < nsteps; ++s) {
    const int cur = s & 1;
    // stage next tile into the other buffer while computing on this one
    if (s + 1 < nsteps)
      async_ld_b128(cur ? loff0 : loff1, gstage + (size_t)(s + 1) * 32);

    // A fragment: lane row m, chunks at hf*8 and 16+hf*8 (ISA 16-bit A layout)
    const bf16_t* ar = arow + s * 32;
    v16bf af = cat8(*(const v8bf*)(ar + hf * 8), *(const v8bf*)(ar + 16 + hf * 8));

#pragma unroll
    for (int j = 0; j < 4; ++j) {
      // B fragment: lane column, chunks at hf*8 and 16+hf*8 (matches A layout)
      const bf16_t* bp = &tileB[cur][(j * 16 + lq) * 32];
      v16bf bfrag = cat8(*(const v8bf*)(bp + hf * 8), *(const v8bf*)(bp + 16 + hf * 8));
      acc[j] = wmma_bf16(af, bfrag, acc[j]);
    }
    wait_async0();     // next tile landed
    __syncthreads();   // everyone done reading cur before it is overwritten
  }

  // epilogue (C layout: lane col = lq, rows r + 8*hf)
#pragma unroll
  for (int j = 0; j < 4; ++j) {
    int n = n0 + j * 16 + lq;
    float bi = bias[n];
#pragma unroll
    for (int r = 0; r < 8; ++r) {
      int m = m0 + r + 8 * hf;
      if (m >= M) continue;
      float val = acc[j][r] + bi;
      if (EPI == EPI_RESID) {
        fout[(size_t)m * Nc + n] += val;
      } else if (EPI == EPI_GELU) {
        float gl = 0.5f * val * (1.0f + erff(val * 0.70710678f));
        bout[(size_t)m * Nc + n] = (bf16_t)gl;
      } else if (EPI == EPI_QK) {
        int b = m / kN, t = m - b * kN;
        int hh = n >> 5, dd = n & 31;
        qkout[(((size_t)b * kH + hh) * qkRows + t) * kHD + dd] = (bf16_t)val;
      } else if (EPI == EPI_VT) {
        int b = m / kN, t = m - b * kN;
        int hh = n >> 5, dd = n & 31;
        vtout[(((size_t)b * kH + hh) * kHD + dd) * kKRows + t] = (bf16_t)val;
      }
    }
  }
}

// flash attention: one wave per (b,h,16-query tile); E^T = K*Q^T, O^T = V^T*P^T
__global__ __launch_bounds__(128) void k_attn(const bf16_t* __restrict__ qb,
                                              const bf16_t* __restrict__ kb,
                                              const bf16_t* __restrict__ vtb,
                                              bf16_t* __restrict__ attno) {
  int gid = blockIdx.x * 4 + (threadIdx.x >> 5);   // exactly B*H*17 waves
  int lane = threadIdx.x & 31, lq = lane & 15, hf = lane >> 4;
  int qt = gid % 17;
  int bh = gid / 17;
  int b = bh >> 3, h = bh & 7;
  int q0 = qt * 16;

  // Q^T B-operand: lane column = query, K = 32 head dims (contiguous 16/half-lane)
  const bf16_t* qr = qb + ((size_t)bh * kQRows + q0 + lq) * kHD + hf * 16;
  v16bf qf = *(const v16bf*)qr;

  const bf16_t* kbase  = kb  + ((size_t)bh * kKRows + lq) * kHD;
  const bf16_t* v0base = vtb + ((size_t)bh * kHD + lq) * kKRows;
  const bf16_t* v1base = vtb + ((size_t)bh * kHD + 16 + lq) * kKRows;

  float m_run = -1e30f, s_run = 0.f;
  v8f acc0 = {}, acc1 = {};

#pragma unroll
  for (int kb32 = 0; kb32 < 9; ++kb32) {
    const int j0 = kb32 * 32;
    v8f e[2];
#pragma unroll
    for (int sub = 0; sub < 2; ++sub) {
      const bf16_t* kr = kbase + (size_t)(j0 + sub * 16) * kHD;
      v16bf kf = *(const v16bf*)(kr + hf * 16);
      v8f z = {};
      e[sub] = wmma_bf16(kf, qf, z);     // E^T tile: lane col = query, row = key
    }
    // mask padded keys (compile-time folded: only last block has any)
    float bm = -1e30f;
    float p0[8], p1[8];
#pragma unroll
    for (int r = 0; r < 8; ++r) {
      float ea = e[0][r], eb = e[1][r];
      if (j0 + 16 + r + 8 > kN - 1) {          // any key in this (r,hf) slot may pad
        int jA = j0 + r + 8 * hf;
        int jB = j0 + 16 + r + 8 * hf;
        ea = (jA < kN) ? ea : -1e30f;
        eb = (jB < kN) ? eb : -1e30f;
      }
      e[0][r] = ea; e[1][r] = eb;
      bm = fmaxf(bm, fmaxf(ea, eb));
    }
    bm = fmaxf(bm, __shfl_xor(bm, 16, 32));
    float m_new = fmaxf(m_run, bm);
    float ps = 0.f;
#pragma unroll
    for (int r = 0; r < 8; ++r) {
      p0[r] = __expf(e[0][r] - m_new);
      p1[r] = __expf(e[1][r] - m_new);
      ps += p0[r] + p1[r];
    }
    ps += __shfl_xor(ps, 16, 32);
    float sc = __expf(m_run - m_new);
    s_run = s_run * sc + ps;
#pragma unroll
    for (int r = 0; r < 8; ++r) { acc0[r] *= sc; acc1[r] *= sc; }
    m_run = m_new;

    // repack P^T into B-operand layout: swap key-halves with partner lane
    v16bf pb;
#pragma unroll
    for (int r = 0; r < 8; ++r) {
      float o0 = __shfl_xor(p0[r], 16, 32);
      float o1 = __shfl_xor(p1[r], 16, 32);
      pb[r]     = (bf16_t)(hf ? o1    : p0[r]);
      pb[r + 8] = (bf16_t)(hf ? p1[r] : o0);
    }
    // V^T A-operand fragments (rows = head dims, K = 32 keys)
    const bf16_t* v0p = v0base + j0;
    const bf16_t* v1p = v1base + j0;
    v16bf vf0 = *(const v16bf*)(v0p + hf * 16);
    v16bf vf1 = *(const v16bf*)(v1p + hf * 16);
    acc0 = wmma_bf16(vf0, pb, acc0);
    acc1 = wmma_bf16(vf1, pb, acc1);
  }

  int q = q0 + lq;
  if (q < kN) {
    float inv = 1.0f / (s_run * 16.0f);   // softmax norm * 1/sqrt(E)
    size_t base = ((size_t)b * kN + q) * kE + h * kHD;
#pragma unroll
    for (int r = 0; r < 8; ++r) {
      int d0 = r + 8 * hf;
      attno[base + d0]      = (bf16_t)(acc0[r] * inv);
      attno[base + 16 + d0] = (bf16_t)(acc1[r] * inv);
    }
  }
}

// mean-pool over N, LayerNorm, linear head -> out[b]
__global__ __launch_bounds__(256) void k_head(const float* __restrict__ x,
                                              const float* __restrict__ g,
                                              const float* __restrict__ be,
                                              const float* __restrict__ hw,
                                              const float* __restrict__ hb,
                                              float* __restrict__ out) {
  __shared__ float sm[8];
  int b = blockIdx.x, e = threadIdx.x;
  float s = 0.f;
  for (int t = 0; t < kN; ++t) s += x[((size_t)b * kN + t) * kE + e];
  float pooled = s * (1.0f / 257.0f);
  float s1 = block_sum256(pooled, sm);
  float s2 = block_sum256(pooled * pooled, sm);
  float mean = s1 * (1.f / 256.f);
  float var  = s2 * (1.f / 256.f) - mean * mean;
  float y = (pooled - mean) * rsqrtf(var + 1e-5f) * g[e] + be[e];
  float d = block_sum256(y * hw[e], sm);
  if (e == 0) out[b] = d + hb[0];
}

// ---------------------------------------------------------------- launch
extern "C" void kernel_launch(void* const* d_in, const int* in_sizes, int n_in,
                              void* d_out, int out_size, void* d_ws, size_t ws_size,
                              hipStream_t stream) {
  (void)in_sizes; (void)n_in; (void)out_size; (void)ws_size;
  const int*   user  = (const int*)  d_in[0];
  const int*   item  = (const int*)  d_in[1];
  const float* utab  = (const float*)d_in[2];
  const float* itab  = (const float*)d_in[3];
  const float* cls   = (const float*)d_in[4];
  const float* pos   = (const float*)d_in[5];
  const float* ln1g  = (const float*)d_in[6];
  const float* ln1b  = (const float*)d_in[7];
  const float* Wq    = (const float*)d_in[8];
  const float* bq    = (const float*)d_in[9];
  const float* Wk    = (const float*)d_in[10];
  const float* bk    = (const float*)d_in[11];
  const float* Wv    = (const float*)d_in[12];
  const float* bv    = (const float*)d_in[13];
  const float* Wo    = (const float*)d_in[14];
  const float* bo    = (const float*)d_in[15];
  const float* ln2g  = (const float*)d_in[16];
  const float* ln2b  = (const float*)d_in[17];
  const float* W1    = (const float*)d_in[18];
  const float* b1    = (const float*)d_in[19];
  const float* W2    = (const float*)d_in[20];
  const float* b2    = (const float*)d_in[21];
  const float* hg    = (const float*)d_in[22];
  const float* hbv   = (const float*)d_in[23];
  const float* hW    = (const float*)d_in[24];
  const float* hbias = (const float*)d_in[25];
  float* out = (float*)d_out;

  char* ws = (char*)d_ws;
  size_t off = 0;
  auto take = [&](size_t bytes) -> void* {
    void* p = ws + off;
    off = (off + bytes + 255) & ~(size_t)255;
    return p;
  };
  const size_t DST = 786432;  // bf16 elems per depth of transposed weights
  bf16_t* WT   = (bf16_t*)take((size_t)kDepth * DST * 2);
  float*  x    = (float*) take((size_t)kM * kE * 4);
  bf16_t* hbuf = (bf16_t*)take((size_t)kM * kE * 2);
  bf16_t* abuf = (bf16_t*)take((size_t)kM * kE * 2);
  bf16_t* ffb  = (bf16_t*)take((size_t)kM * kFF * 2);
  bf16_t* qbuf = (bf16_t*)take((size_t)kB * kH * kQRows * kHD * 2);
  bf16_t* kbuf = (bf16_t*)take((size_t)kB * kH * kKRows * kHD * 2);
  bf16_t* vtb  = (bf16_t*)take((size_t)kB * kH * kHD * kKRows * 2);

  bf16_t* QT  = WT + 0;
  bf16_t* KT  = WT + 65536;
  bf16_t* VT  = WT + 131072;
  bf16_t* OT  = WT + 196608;
  bf16_t* W1T = WT + 262144;
  bf16_t* W2T = WT + 524288;

  // weight convert + transpose (fp32 -> bf16 [n][k])
  k_transp<<<dim3(256,  kDepth), 256, 0, stream>>>(Wq, QT,  256,  256,  65536, (long)DST);
  k_transp<<<dim3(256,  kDepth), 256, 0, stream>>>(Wk, KT,  256,  256,  65536, (long)DST);
  k_transp<<<dim3(256,  kDepth), 256, 0, stream>>>(Wv, VT,  256,  256,  65536, (long)DST);
  k_transp<<<dim3(256,  kDepth), 256, 0, stream>>>(Wo, OT,  256,  256,  65536, (long)DST);
  k_transp<<<dim3(1024, kDepth), 256, 0, stream>>>(W1, W1T, 256, 1024, 262144, (long)DST);
  k_transp<<<dim3(1024, kDepth), 256, 0, stream>>>(W2, W2T, 1024, 256, 262144, (long)DST);

  // embeddings + positions
  k_embed<<<(kM * kE) / 256, 256, 0, stream>>>(user, item, utab, itab, cls, pos, x);

  const int gmx = (kM + 127) / 128;   // 129
  for (int d = 0; d < kDepth; ++d) {
    bf16_t* qt  = QT  + (size_t)d * DST;
    bf16_t* kt  = KT  + (size_t)d * DST;
    bf16_t* vt  = VT  + (size_t)d * DST;
    bf16_t* ot  = OT  + (size_t)d * DST;
    bf16_t* w1t = W1T + (size_t)d * DST;
    bf16_t* w2t = W2T + (size_t)d * DST;

    k_ln<<<kM, 256, 0, stream>>>(x, ln1g + d * kE, ln1b + d * kE, hbuf, kM);
    k_gemm<EPI_QK><<<dim3(gmx, 4), 256, 0, stream>>>(hbuf, qt, bq + d * kE, kM, kE, kE,
                                                     nullptr, nullptr, qbuf, kQRows, nullptr);
    k_gemm<EPI_QK><<<dim3(gmx, 4), 256, 0, stream>>>(hbuf, kt, bk + d * kE, kM, kE, kE,
                                                     nullptr, nullptr, kbuf, kKRows, nullptr);
    k_gemm<EPI_VT><<<dim3(gmx, 4), 256, 0, stream>>>(hbuf, vt, bv + d * kE, kM, kE, kE,
                                                     nullptr, nullptr, nullptr, 0, vtb);
    k_attn<<<(kB * kH * 17) / 4, 128, 0, stream>>>(qbuf, kbuf, vtb, abuf);
    k_gemm<EPI_RESID><<<dim3(gmx, 4), 256, 0, stream>>>(abuf, ot, bo + d * kE, kM, kE, kE,
                                                        x, nullptr, nullptr, 0, nullptr);
    k_ln<<<kM, 256, 0, stream>>>(x, ln2g + d * kE, ln2b + d * kE, hbuf, kM);
    k_gemm<EPI_GELU><<<dim3(gmx, 16), 256, 0, stream>>>(hbuf, w1t, b1 + d * kFF, kM, kE, kFF,
                                                        nullptr, ffb, nullptr, 0, nullptr);
    k_gemm<EPI_RESID><<<dim3(gmx, 4), 256, 0, stream>>>(ffb, w2t, b2 + d * kE, kM, kFF, kE,
                                                        x, nullptr, nullptr, 0, nullptr);
  }
  k_head<<<kB, 256, 0, stream>>>(x, hg, hbv, hW, hbias, out);
}
